// RGCNEntityClassifier_70566312673748
// MI455X (gfx1250) — matrix operand order repaired and checked by reading, not compiled, not run on hardware
//
#include <hip/hip_runtime.h>
#include <hip/hip_bf16.h>

// ---------------------------------------------------------------------------
// RGCN entity classifier forward on gfx1250 (MI455X).
// Dense projections run as bf16 WMMA (V_WMMA_F32_16X16X32_BF16) with the
// per-relation aggregates + self path fused into a single K=1088 GEMM.
// Weights are staged into LDS per-workgroup via the Tensor Data Mover.
// ---------------------------------------------------------------------------

typedef __attribute__((ext_vector_type(16))) __bf16 v16bf;
typedef __attribute__((ext_vector_type(8)))  float  v8f;
typedef __attribute__((ext_vector_type(4)))  unsigned int u32x4;
typedef __attribute__((ext_vector_type(8)))  int          i32x8;
typedef __attribute__((ext_vector_type(4)))  int          i32x4;

constexpr int N_NODES = 50000;
constexpr int N_REL   = 16;
constexpr int N_BASIS = 8;
constexpr int D_IN    = 64;
constexpr int D_HID   = 64;
constexpr int N_CLASS = 16;
constexpr int N_EDGES = 800000;
constexpr int K_REL   = N_REL * D_IN;   // 1024 (per-relation part of K)
constexpr int K_TOT   = K_REL + D_IN;   // 1088 (+ self path)
constexpr int NUM_KT  = K_TOT / 32;     // 34 WMMA K-steps
constexpr int M_TILES = N_NODES / 16;   // 3125 (exact)

// ---------------------------------------------------------------------------
__global__ void zero_f32(float* __restrict__ p, long long n4) {
  long long i = (long long)blockIdx.x * blockDim.x + threadIdx.x;
  long long stride = (long long)gridDim.x * blockDim.x;
  float4 z = make_float4(0.f, 0.f, 0.f, 0.f);
  for (; i < n4; i += stride) reinterpret_cast<float4*>(p)[i] = z;
}

// counts[r, n] = #edges with type r, dst n
__global__ void count_deg(const int* __restrict__ etype, const int* __restrict__ dst,
                          float* __restrict__ counts) {
  int e = blockIdx.x * blockDim.x + threadIdx.x;
  if (e >= N_EDGES) return;
  atomicAdd(&counts[etype[e] * N_NODES + dst[e]], 1.0f);
}

__global__ void inv_deg(const int* __restrict__ etype, const int* __restrict__ dst,
                        const float* __restrict__ counts, float* __restrict__ inv_c) {
  int e = blockIdx.x * blockDim.x + threadIdx.x;
  if (e >= N_EDGES) return;
  float c = counts[etype[e] * N_NODES + dst[e]];
  inv_c[e] = 1.0f / fmaxf(c, 1.0f);
}

// ---------------------------------------------------------------------------
// Build bf16 weight matrix, pre-swizzled into WMMA B-fragment order.
// Virtual stacked weight W[K_TOT, dout]: rows 0..1023 = W_rel (r = K>>6,
// i = K&63, W_rel = sum_b coeffs[r,b]*bases[b,i,:]); rows 1024.. = self.
// Fragment layout (assumed dense 32x16 bf16 B):
//   lane<16 : N = nt*16+lane, elements e=0..15 -> K = kt*32 + e
//   lane>=16: N = nt*16+lane-16,              -> K = kt*32 + 16 + e
// Stored at bsw[((kt*numNt + nt)*32 + lane)*16 + e].
__global__ void build_bsw(const float* __restrict__ bases, const float* __restrict__ coeffs,
                          const float* __restrict__ selfw, __bf16* __restrict__ bsw,
                          int dout) {
  int numNt = dout >> 4;
  int total = NUM_KT * numNt * 512;
  int idx = blockIdx.x * blockDim.x + threadIdx.x;
  if (idx >= total) return;
  int e    = idx & 15;
  int lane = (idx >> 4) & 31;
  int rem  = idx >> 9;            // kt*numNt + nt
  int nt   = rem % numNt;
  int kt   = rem / numNt;
  int K = kt * 32 + ((lane >> 4) << 4) + e;
  int n = nt * 16 + (lane & 15);
  float w;
  if (K < K_REL) {
    int r = K >> 6, i = K & 63;
    float acc = 0.f;
#pragma unroll
    for (int b = 0; b < N_BASIS; ++b)
      acc += coeffs[r * N_BASIS + b] * bases[(b * D_IN + i) * dout + n];
    w = acc;
  } else {
    w = selfw[(K - K_REL) * dout + n];
  }
  bsw[idx] = (__bf16)w;
}

// ---------------------------------------------------------------------------
// Scatter normalized messages into agg[rel, node, 64] with f32 atomics.
// 16 threads per edge, float4 each.
__global__ void scatter_msgs(const int* __restrict__ src, const int* __restrict__ dst,
                             const int* __restrict__ etype, const float* __restrict__ inv_c,
                             const float* __restrict__ feat, float* __restrict__ agg) {
  long long tid = (long long)blockIdx.x * blockDim.x + threadIdx.x;
  if (tid >= (long long)N_EDGES * 16) return;
  int e = (int)(tid >> 4);
  int i = (int)(tid & 15) * 4;
  int s = src[e], d = dst[e], t = etype[e];
  float inv = inv_c[e];
  float4 v = *reinterpret_cast<const float4*>(feat + (long long)s * 64 + i);
  float* a = agg + ((long long)t * N_NODES + d) * 64 + i;
  atomicAdd(a + 0, v.x * inv);
  atomicAdd(a + 1, v.y * inv);
  atomicAdd(a + 2, v.z * inv);
  atomicAdd(a + 3, v.w * inv);
}

// ---------------------------------------------------------------------------
// Stage the swizzled weight matrix (ndw DWORDs) into LDS with the Tensor
// Data Mover: 1D descriptor, 4-byte elements, single tile. Issued by one
// wave; completion via TENSORcnt, then visibility via workgroup barrier.
// This toolchain exposes the 6-arg builtin:
//   (u32x4 g0, i32x8 g1, i32x4 g2, i32x4 g3, i32x8 extra, i32 cpol)
__device__ inline void tdm_stage_weights(const void* gsrc, void* lds_dst, int ndw) {
#if __has_builtin(__builtin_amdgcn_tensor_load_to_lds)
  unsigned long long ga = (unsigned long long)gsrc;
  unsigned lds_off = (unsigned)(uintptr_t)lds_dst;     // low 32 bits = LDS offset
  // D# group 0: count=1 | lds_addr | global_addr[56:0] | type=2 (ISA 8.3)
  u32x4 g0 = { 1u, lds_off, (unsigned)ga,
               (unsigned)((ga >> 32) & 0x1FFFFFFull) | 0x80000000u };
  // D# group 1 (ISA 8.4): data_size=2 (4B), tensor_dim0 = tile_dim0 = ndw,
  // tensor_dim1 = 1, tile_dim1 = 1, strides = ndw / 0.
  i32x8 g1 = { (int)(2u << 16),
               (int)((unsigned)(ndw & 0xFFFF) << 16),
               (int)(((unsigned)ndw >> 16) | (1u << 16)),
               (int)((unsigned)ndw << 16),
               1,
               ndw,
               0, 0 };
  i32x4 gz4 = { 0, 0, 0, 0 };
  i32x8 gz8 = { 0, 0, 0, 0, 0, 0, 0, 0 };
  __builtin_amdgcn_tensor_load_to_lds(g0, g1, gz4, gz4, gz8, 0);
  __builtin_amdgcn_s_wait_tensorcnt(0);
#else
  // Fallback: cooperative vector copy (no TDM builtin on this toolchain).
  const int4* s = (const int4*)gsrc;
  int4* d = (int4*)lds_dst;
  for (int i = threadIdx.x; i < ndw / 4; i += blockDim.x) d[i] = s[i];
#endif
}

// ---------------------------------------------------------------------------
// Fused GEMM: out[n, :16*NUMNT] = [agg(:,n,:) | feat(n,:)] (1x1088) @ Wsw.
// One wave per 16-node M-tile. K loop is organized per relation r: kt = 2r
// and 2r+1 together consume the contiguous 64-float agg row agg[r][node][:],
// so addressing is a single pointer bumped by N_NODES*64 per relation.
// A-fragment VGPR layout (16-bit A 16x32, contiguous k order used here, same as bsw B):
//   lane<16 : M = lane,    elems 0..15 -> K kt*32+0..15
//   lane>=16: M = lane-16, elems 0..15 -> K kt*32+16..31
// agg loads are non-temporal (read-once; keep L2 for the scatter atomics).
// B fragments come from LDS (staged once per workgroup by the TDM).
template <int NUMNT, bool RELU>
__global__ __launch_bounds__(256)
void rgcn_gemm(const float* __restrict__ agg, const float* __restrict__ feat,
               const __bf16* __restrict__ bsw, float* __restrict__ out) {
  extern __shared__ __align__(16) char smem[];
  __bf16* lds_b = reinterpret_cast<__bf16*>(smem);
  constexpr int BDW = NUM_KT * NUMNT * 256;     // weight DWORD count

  const int lane  = threadIdx.x & 31;
  const int wid   = threadIdx.x >> 5;
  const int mtile = blockIdx.x * 8 + wid;

  if (wid == 0) tdm_stage_weights(bsw, lds_b, BDW);
  __syncthreads();

  if (mtile < M_TILES) {                        // uniform per wave -> EXEC all-1s
    const int node   = mtile * 16 + (lane & 15);
    const int kshift = (lane < 16) ? 0 : 16;

    v8f acc[NUMNT] = {};
    const float*  ap = agg + (long long)node * 64 + kshift;
    const __bf16* bp = lds_b + lane * 16;
    const long long rstride = (long long)N_NODES * 64;

    for (int r = 0; r < N_REL; ++r) {
      if (r + 1 < N_REL) __builtin_prefetch(ap + rstride, 0, 0);
#pragma unroll
      for (int half = 0; half < 2; ++half) {    // kt = 2r + half
        v8f va0 = __builtin_nontemporal_load(reinterpret_cast<const v8f*>(ap + half * 32));
        v8f va1 = __builtin_nontemporal_load(reinterpret_cast<const v8f*>(ap + half * 32 + 8));
        v16bf afrag;
#pragma unroll
        for (int j = 0; j < 8; ++j) {
          afrag[j]     = (__bf16)va0[j];
          afrag[8 + j] = (__bf16)va1[j];
        }
#pragma unroll
        for (int nt = 0; nt < NUMNT; ++nt) {
          v16bf bfrag = *reinterpret_cast<const v16bf*>(bp + (half * NUMNT + nt) * 512);
          acc[nt] = __builtin_amdgcn_wmma_f32_16x16x32_bf16(
              false, afrag, false, bfrag, (short)0, acc[nt], false, false);
        }
      }
      ap += rstride;
      bp += 2 * NUMNT * 512;
    }

    // self path: kt = 32, 33 consume feat[node][:]
    const float* fp = feat + (long long)node * 64 + kshift;
#pragma unroll
    for (int half = 0; half < 2; ++half) {
      v8f va0 = *reinterpret_cast<const v8f*>(fp + half * 32);
      v8f va1 = *reinterpret_cast<const v8f*>(fp + half * 32 + 8);
      v16bf afrag;
#pragma unroll
      for (int j = 0; j < 8; ++j) {
        afrag[j]     = (__bf16)va0[j];
        afrag[8 + j] = (__bf16)va1[j];
      }
#pragma unroll
      for (int nt = 0; nt < NUMNT; ++nt) {
        v16bf bfrag = *reinterpret_cast<const v16bf*>(bp + (half * NUMNT + nt) * 512);
        acc[nt] = __builtin_amdgcn_wmma_f32_16x16x32_bf16(
            false, afrag, false, bfrag, (short)0, acc[nt], false, false);
      }
    }

    // C/D layout: VGPR v -> M = v (lanes 0-15) or v+8 (lanes 16-31), N = lane&15
    const int rbase = (lane < 16) ? 0 : 8;
    constexpr int DOUT = NUMNT * 16;
#pragma unroll
    for (int nt = 0; nt < NUMNT; ++nt) {
#pragma unroll
      for (int v = 0; v < 8; ++v) {
        float x = acc[nt][v];
        if (RELU) x = fmaxf(x, 0.f);
        out[(long long)(mtile * 16 + rbase + v) * DOUT + nt * 16 + (lane & 15)] = x;
      }
    }
  }
}

// ---------------------------------------------------------------------------
extern "C" void kernel_launch(void* const* d_in, const int* in_sizes, int n_in,
                              void* d_out, int out_size, void* d_ws, size_t ws_size,
                              hipStream_t stream) {
  (void)in_sizes; (void)n_in; (void)out_size; (void)ws_size;
  const float* x       = (const float*)d_in[0];
  const float* bases1  = (const float*)d_in[1];
  const float* coeffs1 = (const float*)d_in[2];
  const float* self1   = (const float*)d_in[3];
  const float* bases2  = (const float*)d_in[4];
  const float* coeffs2 = (const float*)d_in[5];
  const float* self2   = (const float*)d_in[6];
  const int* edge_index = (const int*)d_in[7];
  const int* edge_type  = (const int*)d_in[8];
  const int* src = edge_index;                // edge_index[0, :]
  const int* dst = edge_index + N_EDGES;      // edge_index[1, :]

  char* ws = (char*)d_ws;
  size_t off = 0;
  auto carve = [&](size_t bytes) -> char* {
    char* p = ws + off;
    off += (bytes + 255) & ~(size_t)255;
    return p;
  };
  float*  counts = (float*)carve(sizeof(float) * N_REL * N_NODES);              // 3.2 MB
  float*  inv_c  = (float*)carve(sizeof(float) * N_EDGES);                      // 3.2 MB
  float*  agg    = (float*)carve(sizeof(float) * (size_t)N_REL * N_NODES * 64); // 204.8 MB
  float*  h      = (float*)carve(sizeof(float) * (size_t)N_NODES * D_HID);      // 12.8 MB
  __bf16* bsw1   = (__bf16*)carve(sizeof(__bf16) * (size_t)NUM_KT * 4 * 512);   // 139 KB
  __bf16* bsw2   = (__bf16*)carve(sizeof(__bf16) * (size_t)NUM_KT * 1 * 512);   // 35 KB

  const int TB = 256;
  // normalization
  zero_f32<<<2048, TB, 0, stream>>>(counts, (long long)(N_REL * N_NODES) / 4);
  count_deg<<<(N_EDGES + TB - 1) / TB, TB, 0, stream>>>(edge_type, dst, counts);
  inv_deg<<<(N_EDGES + TB - 1) / TB, TB, 0, stream>>>(edge_type, dst, counts, inv_c);
  // pre-swizzled bf16 weights
  build_bsw<<<(NUM_KT * 4 * 512 + TB - 1) / TB, TB, 0, stream>>>(bases1, coeffs1, self1, bsw1, D_HID);
  build_bsw<<<(NUM_KT * 1 * 512 + TB - 1) / TB, TB, 0, stream>>>(bases2, coeffs2, self2, bsw2, N_CLASS);

  const long long aggN4 = (long long)N_REL * N_NODES * 64 / 4;
  const int scatterBlocks = (int)(((long long)N_EDGES * 16 + TB - 1) / TB);
  const int gemmBlocks = (M_TILES + 7) / 8;
  const size_t lds1 = sizeof(__bf16) * (size_t)NUM_KT * 4 * 512;   // 139264 B
  const size_t lds2 = sizeof(__bf16) * (size_t)NUM_KT * 1 * 512;   //  34816 B

  // layer 1
  zero_f32<<<8192, TB, 0, stream>>>(agg, aggN4);
  scatter_msgs<<<scatterBlocks, TB, 0, stream>>>(src, dst, edge_type, inv_c, x, agg);
  rgcn_gemm<4, true><<<gemmBlocks, TB, lds1, stream>>>(agg, x, bsw1, h);

  // layer 2 (reuse agg buffer)
  zero_f32<<<8192, TB, 0, stream>>>(agg, aggN4);
  scatter_msgs<<<scatterBlocks, TB, 0, stream>>>(src, dst, edge_type, inv_c, h, agg);
  rgcn_gemm<1, false><<<gemmBlocks, TB, lds2, stream>>>(agg, h, bsw2, (float*)d_out);
}
